// SparseAutoencoder_17274358465205
// MI455X (gfx1250) — matrix operand order repaired and checked by reading, not compiled, not run on hardware
//
#include <hip/hip_runtime.h>
#include <hip/hip_bf16.h>

// ---------------------------------------------------------------------------
// FoldingNet autoencoder for MI455X (gfx1250): f16 WMMA GEMMs + scalar glue.
// Wave tile 32x64 (8 WMMA per 12 b128 loads), block tile 128x128.
// ---------------------------------------------------------------------------

typedef __attribute__((ext_vector_type(16))) _Float16 v16h;
typedef __attribute__((ext_vector_type(8)))  float    v8f;

#define B_   16
#define N_   2048
#define D_   4
#define G_   45
#define M_   (G_ * G_)      // 2025
#define CW_  512
#define H1_  64
#define H2_  128
#define FH_  512
#define KP_  544            // 514/515 padded up to a multiple of 32
#define RENC (B_ * N_)      // 32768 encoder rows
#define RDEC (B_ * M_)      // 32400 decoder rows

union FragU { v16h h; uint4 u[2]; };

// A fragment: 16x32 f16, row-major source, ld in halfs (multiple of 8).
// ISA layout: lanes 0-15 row M=lane, K in {0..7} then {16..23};
//             lanes 16-31 row M=lane-16, K in {8..15} then {24..31}.
__device__ inline v16h load_a_frag(const _Float16* base, int ld, int kofs) {
    int lane = threadIdx.x & 31;
    int r    = lane & 15;
    int sel  = (lane >> 4) & 1;            // 0 or 1
    const uint4* p = reinterpret_cast<const uint4*>(base + (size_t)r * ld + kofs);
    FragU f;
    f.u[0] = p[sel];                        // K = sel*8 .. sel*8+7
    f.u[1] = p[sel + 2];                    // K = sel*8+16 .. sel*8+23
    return f.h;
}

// B fragment: 32x16 f16 loaded from transposed weights Wt[n][k] (N x Kpad).
// ISA (column-striped): lanes 0-15 col N=lane, K in {0..7} then {16..23};
//                       lanes 16-31 col N=lane-16, K in {8..15} then {24..31}.
__device__ inline v16h load_b_frag(const _Float16* baseT, int ld, int kofs) {
    int lane = threadIdx.x & 31;
    int c    = lane & 15;
    int k0   = (lane >> 4) << 3;           // 0 or 8
    const uint4* p = reinterpret_cast<const uint4*>(baseT + (size_t)c * ld + kofs + k0);
    FragU f;
    f.u[0] = p[0];
    f.u[1] = p[2];
    return f.h;
}

// Generic GEMM: C[f16] = act(A[f16, MxK] * Wt^T[f16, NxK] + bias[f32, N])
// Block = 256 threads = 8 waves (4 row x 2 col); wave tile 32x64.
__global__ void __launch_bounds__(256) wmma_gemm_bias(
    const _Float16* __restrict__ A, int lda,
    const _Float16* __restrict__ Bt, int ldb,
    const float* __restrict__ bias,
    _Float16* __restrict__ C, int ldc,
    int Mrows, int Ktiles, int doRelu)
{
    const int wave    = threadIdx.x >> 5;
    const int rowBase = blockIdx.x * 128 + (wave & 3) * 32;
    const int colBase = blockIdx.y * 128 + (wave >> 2) * 64;
    if (rowBase >= Mrows) return;            // wave-uniform guard
    const bool twoRows = (rowBase + 16) < Mrows;

    const _Float16* Ab0 = A + (size_t)rowBase * lda;
    const _Float16* Ab1 = Ab0 + (size_t)(twoRows ? 16 : 0) * lda;
    const _Float16* Bb[4] = {
        Bt + (size_t)(colBase +  0) * ldb,
        Bt + (size_t)(colBase + 16) * ldb,
        Bt + (size_t)(colBase + 32) * ldb,
        Bt + (size_t)(colBase + 48) * ldb,
    };

    v8f acc[8] = {};                         // [0..3]: rows 0-15, [4..7]: rows 16-31
    for (int kt = 0; kt < Ktiles; ++kt) {
        const int k = kt * 32;
        v16h a0 = load_a_frag(Ab0, lda, k);
        v16h a1 = load_a_frag(Ab1, lda, k);
#pragma unroll
        for (int j = 0; j < 4; ++j) {
            v16h b = load_b_frag(Bb[j], ldb, k);
            acc[j] = __builtin_amdgcn_wmma_f32_16x16x32_f16(
                false, a0, false, b, (short)0, acc[j], false, false);
            acc[j + 4] = __builtin_amdgcn_wmma_f32_16x16x32_f16(
                false, a1, false, b, (short)0, acc[j + 4], false, false);
        }
    }

    // Epilogue: C/D layout — VGPR i: lanes 0-15 M=i, lanes 16-31 M=i+8.
    const int lane = threadIdx.x & 31;
    const int cn   = lane & 15;
    const int rOff = (lane >> 4) ? 8 : 0;
#pragma unroll
    for (int j = 0; j < 4; ++j) {
        const int col = colBase + j * 16 + cn;
        const float bv = bias[col];
#pragma unroll
        for (int i = 0; i < 8; ++i) {
            float v = acc[j][i] + bv;
            if (doRelu) v = v > 0.f ? v : 0.f;
            const int row = rowBase + rOff + i;
            C[(size_t)row * ldc + col] = (_Float16)v;
        }
        if (twoRows) {
#pragma unroll
            for (int i = 0; i < 8; ++i) {
                float v = acc[j + 4][i] + bv;
                if (doRelu) v = v > 0.f ? v : 0.f;
                const int row = rowBase + 16 + rOff + i;
                C[(size_t)row * ldc + col] = (_Float16)v;
            }
        }
    }
}

// Transpose + f16-convert weights: W (Kin x Nout, f32) -> Wt (Nout x Kpad, f16).
__global__ void conv_weight_t(const float* __restrict__ W, int Kin, int Nout,
                              int Kpad, _Float16* __restrict__ Wt)
{
    int idx = blockIdx.x * blockDim.x + threadIdx.x;
    if (idx >= Nout * Kpad) return;
    int n = idx / Kpad, k = idx % Kpad;
    float v = (k < Kin) ? W[(size_t)k * Nout + n] : 0.f;
    Wt[idx] = (_Float16)v;
}

// Encoder layer 1 (K=4, too small for WMMA): h1 = relu(points@ew1 + eb1), f16.
__global__ void enc_layer1(const float* __restrict__ pts,
                           const float* __restrict__ w,
                           const float* __restrict__ b,
                           _Float16* __restrict__ out)
{
    int idx = blockIdx.x * blockDim.x + threadIdx.x;
    if (idx >= RENC * H1_) return;
    int r = idx >> 6, c = idx & 63;
    const float* p = pts + (size_t)r * D_;
    float s = b[c];
#pragma unroll
    for (int k = 0; k < D_; ++k) s += p[k] * w[k * H1_ + c];
    out[idx] = (_Float16)(s > 0.f ? s : 0.f);
}

// Max-pool over N per batch: theta[b][c] = max_n h3[b*N+n][c].
__global__ void maxpool_theta(const _Float16* __restrict__ h3,
                              float* __restrict__ theta)
{
    int idx = blockIdx.x * blockDim.x + threadIdx.x;
    if (idx >= B_ * CW_) return;
    int bb = idx >> 9, c = idx & 511;
    const _Float16* p = h3 + (size_t)bb * N_ * CW_ + c;
    float m = -3.4e38f;
    for (int n = 0; n < N_; ++n) {
        float v = (float)p[(size_t)n * CW_];
        m = v > m ? v : m;
    }
    theta[idx] = m;
}

// Build folding input Xf (RDEC x KP_, f16): [theta | grid-or-fold1 | zero pad].
__global__ void build_xf(const float* __restrict__ theta,
                         const float* __restrict__ grid,
                         const float* __restrict__ fold1,   // nullptr -> use grid
                         _Float16* __restrict__ Xf)
{
    long long idx = (long long)blockIdx.x * blockDim.x + threadIdx.x;
    if (idx >= (long long)RDEC * KP_) return;
    int r = (int)(idx / KP_), k = (int)(idx % KP_);
    int bb = r / M_, m = r % M_;
    float v = 0.f;
    if (k < CW_) {
        v = theta[bb * CW_ + k];
    } else if (fold1 == nullptr) {
        if (k < CW_ + 2) v = grid[m * 2 + (k - CW_)];
    } else {
        if (k < CW_ + 3) v = fold1[(size_t)r * 3 + (k - CW_)];
    }
    Xf[idx] = (_Float16)v;
}

// Final 512->3 head (N=3 too small for WMMA): out = Y@W + b, f32.
__global__ void final3(const _Float16* __restrict__ Y,
                       const float* __restrict__ W,
                       const float* __restrict__ b3,
                       float* __restrict__ out, int R)
{
    int idx = blockIdx.x * blockDim.x + threadIdx.x;
    if (idx >= R * 3) return;
    int r = idx / 3, j = idx % 3;
    const _Float16* y = Y + (size_t)r * FH_;
    float s = b3[j];
    for (int k = 0; k < FH_; ++k) s += (float)y[k] * W[k * 3 + j];
    out[idx] = s;
}

__global__ void zero_acc(float* acc) {
    if (threadIdx.x < 2) acc[threadIdx.x] = 0.f;
}

// Chamfer direction A: per input point, min over decoded points; sum via atomics.
__global__ void chamfer_a(const float* __restrict__ pts,
                          const float* __restrict__ pc,
                          float* __restrict__ acc)
{
    int idx = blockIdx.x * blockDim.x + threadIdx.x;
    if (idx >= B_ * N_) return;
    int bb = idx >> 11;
    const float* p = pts + (size_t)idx * D_;
    float ax = p[0], ay = p[1], az = p[2] + p[3];
    const float* q = pc + (size_t)bb * M_ * 3;
    float mn = 3.4e38f;
    for (int m = 0; m < M_; ++m) {
        float dx = ax - q[m * 3 + 0];
        float dy = ay - q[m * 3 + 1];
        float dz = az - q[m * 3 + 2];
        float d = dx * dx + dy * dy + dz * dz;
        mn = d < mn ? d : mn;
    }
    atomicAdd(&acc[0], mn);
}

// Chamfer direction B: per decoded point, min over input points.
__global__ void chamfer_b(const float* __restrict__ pts,
                          const float* __restrict__ pc,
                          float* __restrict__ acc)
{
    int idx = blockIdx.x * blockDim.x + threadIdx.x;
    if (idx >= B_ * M_) return;
    int bb = idx / M_;
    const float* q = pc + (size_t)idx * 3;
    float bx = q[0], by = q[1], bz = q[2];
    const float* p = pts + (size_t)bb * N_ * D_;
    float mn = 3.4e38f;
    for (int n = 0; n < N_; ++n) {
        float dx = bx - p[n * 4 + 0];
        float dy = by - p[n * 4 + 1];
        float dz = bz - (p[n * 4 + 2] + p[n * 4 + 3]);
        float d = dx * dx + dy * dy + dz * dz;
        mn = d < mn ? d : mn;
    }
    atomicAdd(&acc[1], mn);
}

__global__ void finalize_loss(const float* __restrict__ acc, float* __restrict__ out) {
    if (threadIdx.x == 0)
        out[0] = acc[0] / (float)(B_ * N_) + acc[1] / (float)(B_ * M_);
}

// ---------------------------------------------------------------------------

extern "C" void kernel_launch(void* const* d_in, const int* in_sizes, int n_in,
                              void* d_out, int out_size, void* d_ws, size_t ws_size,
                              hipStream_t stream) {
    const float* points = (const float*)d_in[0];
    const float* ew1  = (const float*)d_in[1];
    const float* eb1  = (const float*)d_in[2];
    const float* ew2  = (const float*)d_in[3];
    const float* eb2  = (const float*)d_in[4];
    const float* ew3  = (const float*)d_in[5];
    const float* eb3  = (const float*)d_in[6];
    const float* f1w1 = (const float*)d_in[7];
    const float* f1b1 = (const float*)d_in[8];
    const float* f1w2 = (const float*)d_in[9];
    const float* f1b2 = (const float*)d_in[10];
    const float* f1w3 = (const float*)d_in[11];
    const float* f1b3 = (const float*)d_in[12];
    const float* f2w1 = (const float*)d_in[13];
    const float* f2b1 = (const float*)d_in[14];
    const float* f2w2 = (const float*)d_in[15];
    const float* f2b2 = (const float*)d_in[16];
    const float* f2w3 = (const float*)d_in[17];
    const float* f2b3 = (const float*)d_in[18];
    const float* grid = (const float*)d_in[19];

    float* theta  = (float*)d_out;                       // B*CW = 8192
    float* pc_out = (float*)d_out + B_ * CW_;            // B*M*3 = 97200
    float* loss   = (float*)d_out + B_ * CW_ + RDEC * 3; // 1

    // Workspace carve-up (256B aligned).
    char* ws = (char*)d_ws;
    size_t off = 0;
    auto take = [&](size_t bytes) -> char* {
        char* p = ws + off;
        off = (off + bytes + 255) & ~(size_t)255;
        return p;
    };
    _Float16* W2t   = (_Float16*)take((size_t)H2_ * H1_ * 2);
    _Float16* W3t   = (_Float16*)take((size_t)CW_ * H2_ * 2);
    _Float16* F1W1t = (_Float16*)take((size_t)FH_ * KP_ * 2);
    _Float16* F1W2t = (_Float16*)take((size_t)FH_ * FH_ * 2);
    _Float16* F2W1t = (_Float16*)take((size_t)FH_ * KP_ * 2);
    _Float16* F2W2t = (_Float16*)take((size_t)FH_ * FH_ * 2);
    _Float16* A1    = (_Float16*)take((size_t)RENC * H1_ * 2);
    _Float16* A2    = (_Float16*)take((size_t)RENC * H2_ * 2);
    _Float16* XF    = (_Float16*)take((size_t)RDEC * KP_ * 2);   // also h3 (RENC*CW fits)
    _Float16* Y1    = (_Float16*)take((size_t)RDEC * FH_ * 2);
    _Float16* Y2    = (_Float16*)take((size_t)RDEC * FH_ * 2);
    float*    fold1 = (float*)take((size_t)RDEC * 3 * 4);
    float*    acc   = (float*)take(2 * 4);
    _Float16* H3    = XF;  // pre-pool encoder output reuses the big buffer

    const int T = 256;
    auto blocks = [](long long n, int t) { return (unsigned)((n + t - 1) / t); };

    // 0) weight prep + loss accumulator reset
    conv_weight_t<<<blocks((long long)H2_ * H1_, T), T, 0, stream>>>(ew2, H1_, H2_, H1_, W2t);
    conv_weight_t<<<blocks((long long)CW_ * H2_, T), T, 0, stream>>>(ew3, H2_, CW_, H2_, W3t);
    conv_weight_t<<<blocks((long long)FH_ * KP_, T), T, 0, stream>>>(f1w1, CW_ + 2, FH_, KP_, F1W1t);
    conv_weight_t<<<blocks((long long)FH_ * FH_, T), T, 0, stream>>>(f1w2, FH_, FH_, FH_, F1W2t);
    conv_weight_t<<<blocks((long long)FH_ * KP_, T), T, 0, stream>>>(f2w1, CW_ + 3, FH_, KP_, F2W1t);
    conv_weight_t<<<blocks((long long)FH_ * FH_, T), T, 0, stream>>>(f2w2, FH_, FH_, FH_, F2W2t);
    zero_acc<<<1, 32, 0, stream>>>(acc);

    // 1) encoder
    enc_layer1<<<blocks((long long)RENC * H1_, T), T, 0, stream>>>(points, ew1, eb1, A1);
    wmma_gemm_bias<<<dim3(RENC / 128, H2_ / 128), T, 0, stream>>>(
        A1, H1_, W2t, H1_, eb2, A2, H2_, RENC, H1_ / 32, 1);
    wmma_gemm_bias<<<dim3(RENC / 128, CW_ / 128), T, 0, stream>>>(
        A2, H2_, W3t, H2_, eb3, H3, CW_, RENC, H2_ / 32, 0);
    maxpool_theta<<<blocks(B_ * CW_, T), T, 0, stream>>>(H3, theta);

    // 2) folding 1
    const unsigned gxDec = (RDEC + 127) / 128;  // 254 (tail block wave-guarded)
    build_xf<<<blocks((long long)RDEC * KP_, T), T, 0, stream>>>(theta, grid, nullptr, XF);
    wmma_gemm_bias<<<dim3(gxDec, FH_ / 128), T, 0, stream>>>(
        XF, KP_, F1W1t, KP_, f1b1, Y1, FH_, RDEC, KP_ / 32, 1);
    wmma_gemm_bias<<<dim3(gxDec, FH_ / 128), T, 0, stream>>>(
        Y1, FH_, F1W2t, FH_, f1b2, Y2, FH_, RDEC, FH_ / 32, 1);
    final3<<<blocks((long long)RDEC * 3, T), T, 0, stream>>>(Y2, f1w3, f1b3, fold1, RDEC);

    // 3) folding 2
    build_xf<<<blocks((long long)RDEC * KP_, T), T, 0, stream>>>(theta, grid, fold1, XF);
    wmma_gemm_bias<<<dim3(gxDec, FH_ / 128), T, 0, stream>>>(
        XF, KP_, F2W1t, KP_, f2b1, Y1, FH_, RDEC, KP_ / 32, 1);
    wmma_gemm_bias<<<dim3(gxDec, FH_ / 128), T, 0, stream>>>(
        Y1, FH_, F2W2t, FH_, f2b2, Y2, FH_, RDEC, FH_ / 32, 1);
    final3<<<blocks((long long)RDEC * 3, T), T, 0, stream>>>(Y2, f2w3, f2b3, pc_out, RDEC);

    // 4) chamfer loss
    chamfer_a<<<blocks(B_ * N_, T), T, 0, stream>>>(points, pc_out, acc);
    chamfer_b<<<blocks(B_ * M_, T), T, 0, stream>>>(points, pc_out, acc);
    finalize_loss<<<1, 32, 0, stream>>>(acc, loss);
}
